// unidirectional_36344013258822
// MI455X (gfx1250) — matrix operand order repaired and checked by reading, not compiled, not run on hardware
//
#include <hip/hip_runtime.h>

// Problem constants (from the reference): B=8, C=1, H=1024, L=4096
#define Bn 8
#define Hn 1024
#define Ln 4096
#define BM 128     // output timesteps per workgroup (8 waves x 16 rows)
#define SC 128     // K (input-time) chunk staged in LDS per iteration

typedef __attribute__((ext_vector_type(16))) _Float16 v16h;
typedef __attribute__((ext_vector_type(4)))  _Float16 v4h;
typedef __attribute__((ext_vector_type(8)))  float    v8f;

// Causal long conv as Toeplitz-GEMM on WMMA:
//   y[b,h,t] = sum_{s<=t} k[h,t-s] * x[b,h,s] + D[h]*x[b,h,t]
// Per head h: Y(L x 8) = T_k(L x L, lower-tri Toeplitz) * X(L x 8),
// tiled 16x16x32 f16 WMMA, f32 accumulate, f32 residual epilogue.
__global__ __launch_bounds__(256, 1)
void ssm_toeplitz_wmma(const float* __restrict__ x,
                       const float* __restrict__ k,
                       const float* __restrict__ Dv,
                       float* __restrict__ out)
{
    // LDS: 16 x-columns (rows 8..15 permanently zero -> padded batch columns
    // need no per-element masking in the hot loop) + reversed-k window. 4.5 KB.
    __shared__ _Float16 xs[16][SC];       // xs[n][s - s0]; n>=8 stays zero
    __shared__ _Float16 kr[BM + SC];      // kr[i] = k[h, (T0 - s0) + 127 - i], 0 if lag<0

    const int mc   = blockIdx.x;          // 0..31  : which 128-row output chunk
    const int h    = blockIdx.y;          // 0..1023: head
    const int T0   = mc * BM;
    const int tid  = threadIdx.x;
    const int lane = tid & 31;            // wave32
    const int w    = tid >> 5;            // wave id 0..7 -> rows [T0+16w, T0+16w+16)
    const int hs   = lane >> 4;           // half-wave select (ISA 16-bit fragment layouts)
    const int nlo  = lane & 15;           // A: row m; B/D: column n

    // Zero the padded batch columns once; never touched again.
    for (int i = tid; i < Bn * SC; i += 256) {
        xs[Bn + (i >> 7)][i & (SC - 1)] = (_Float16)0.0f;
    }

    // Staging geometry: wave w owns batch row b=w; lane covers 4 contiguous s.
    const size_t xrow = ((size_t)w * Hn + h) * Ln;   // x[b=w, h, :]
    const int    sx   = lane << 2;                   // 4 contiguous timesteps

    v8f acc = {};                         // 16x16 f32 accumulator (8 VGPRs)

    const int nchunks = mc + 1;           // causal: s0 ranges [0, T0] step SC
    for (int c = 0; c < nchunks; ++c) {
        const int s0 = c * SC;

        // Prefetch next x chunk toward L2/L0 (global_prefetch_b8).
        if (c + 1 < nchunks) {
            __builtin_prefetch(&x[xrow + (size_t)(s0 + SC) + (size_t)sx], 0, 1);
        }

        // ---- Stage x[b=w, h, s0..s0+SC) as f16: one float4 load + ds_store_b64 ----
        {
            const float4 xv = *(const float4*)&x[xrow + (size_t)(s0 + sx)];
            v4h xh;
            xh[0] = (_Float16)xv.x; xh[1] = (_Float16)xv.y;
            xh[2] = (_Float16)xv.z; xh[3] = (_Float16)xv.w;
            *(v4h*)&xs[w][sx] = xh;
        }
        // ---- Stage reversed kernel window: kr[i] = k[h, (T0-s0)+127-i] ----
        // Thread t (t<64) covers kr[4t..4t+3] from k[q..q+3] reversed, where
        // q = (T0-s0)+124-4t. q = 0 mod 4, so either all 4 lags valid or all < 0.
        if (tid < 64) {
            const int q = (T0 - s0) + (BM - 4) - (tid << 2);
            v4h kh;
            if (q >= 0) {
                const float4 kv = *(const float4*)&k[(size_t)h * Ln + q];
                kh[0] = (_Float16)kv.w; kh[1] = (_Float16)kv.z;
                kh[2] = (_Float16)kv.y; kh[3] = (_Float16)kv.x;
            } else {
                kh[0] = kh[1] = kh[2] = kh[3] = (_Float16)0.0f;
            }
            *(v4h*)&kr[tid << 2] = kh;
        }
        __syncthreads();

        // ---- 4 WMMA K-steps of 32 over this chunk ----
#pragma unroll
        for (int j = 0; j < 4; ++j) {
            v16h a, bf;
            // A (16x32 Toeplitz tile): A[m][kk] = k[(T0+16w+m) - (s0+32j+kk)]
            //                                  = kr[(127 - 16w - m) + 32j + kk]
            const int abase = (BM - 1) - (16 * w + nlo) + 32 * j;
#pragma unroll
            for (int v = 0; v < 16; ++v) {
                // A layout (matches B): lanes 0-15 hold K 0..15, lanes 16-31 hold K 16..31
                const int kkA = (hs << 4) + v;
                a[v] = kr[abase + kkA];
                // B layout (matches A): lanes 0-15 hold K 0..15, lanes 16-31 hold K 16..31
                // B (32x16): B[kk][n] = x[b=n, h, s0 + 32j + kk]; columns n>=8 read LDS zeros.
                bf[v] = xs[nlo][32 * j + (hs << 4) + v];
            }
            acc = __builtin_amdgcn_wmma_f32_16x16x32_f16(
                /*neg_a=*/false, a, /*neg_b=*/false, bf,
                /*c_mod=*/(short)0, acc, /*reuse_a=*/false, /*reuse_b=*/false);
        }
        __syncthreads();   // protect LDS before next chunk's staging
    }

    // ---- Epilogue: f32 residual + store. D layout: VGPR r -> M = r + 8*hs, N = nlo ----
    if (nlo < Bn) {
        const float Dh = Dv[h];
        const size_t rowbase = ((size_t)nlo * Hn + h) * Ln;
#pragma unroll
        for (int r = 0; r < 8; ++r) {
            const int t = T0 + 16 * w + (r + (hs << 3));
            const float xv = x[rowbase + t];
            out[rowbase + t] = acc[r] + xv * Dh;
        }
    }
}

extern "C" void kernel_launch(void* const* d_in, const int* in_sizes, int n_in,
                              void* d_out, int out_size, void* d_ws, size_t ws_size,
                              hipStream_t stream) {
    (void)in_sizes; (void)n_in; (void)d_ws; (void)ws_size; (void)out_size;
    const float* x  = (const float*)d_in[0];   // (B, H, L) f32
    const float* kk = (const float*)d_in[1];   // (1, H, L) f32
    const float* Dv = (const float*)d_in[2];   // (1, H, 1) f32
    float* out = (float*)d_out;                // (B, H, L) f32

    dim3 grid(Ln / BM, Hn);                    // 32 x 1024 workgroups
    ssm_toeplitz_wmma<<<grid, 256, 0, stream>>>(x, kk, Dv, out);
}
